// LundNet_33423435497558
// MI455X (gfx1250) — matrix operand order repaired and checked by reading, not compiled, not run on hardware
//
#include <hip/hip_runtime.h>

// ---------------------------------------------------------------------------
// LundNet-style EdgeConv GNN for MI455X (gfx1250, wave32, WMMA).
//
// Heavy math (edge MLPs, shortcut linears, seq1) runs as bf16 WMMA GEMMs
// (v_wmma_f32_16x16x32_bf16) with fp32 accumulation. Edge gather is fused
// into the A-fragment build of GEMM1 (node features stay L2-resident);
// BN(affine)+ReLU of the MLP hidden layer is fused into GEMM2's A build.
// BatchNorm biases before BN are dropped (they cancel in (y-mean)).
//
// Workspace layout (~1.27 GB):
//   Y1  bf16 E*128      Y2 bf16 E*128      Hc f32 N*448 (x1..x6 concat)
//   Hb  bf16 N*448      Xb bf16 N*128      S  bf16 N*128   H1 bf16 N*384
//   accum f32 N*128     cnt f32 N          gpool f32 B*384  gcnt f32 B
//   stats: 3 slots x {sum,ssq,scale,shift}[512] f32
// ---------------------------------------------------------------------------

#define GN 100000
#define GE 1600000
#define GB 256
#define BNEPS 1e-5f

typedef __attribute__((ext_vector_type(16))) __bf16 v16bf;
typedef __attribute__((ext_vector_type(8)))  float  v8f;

union Frag16 { v16bf v; uint4 q[2]; __bf16 h[16]; unsigned int u[8]; };

__device__ __forceinline__ void atomAddF(float* p, float v) {
    __hip_atomic_fetch_add(p, v, __ATOMIC_RELAXED, __HIP_MEMORY_SCOPE_AGENT);
}
__device__ __forceinline__ float bfbits2f(unsigned int ubits16) {
    unsigned int x = ubits16 << 16;
    float f; __builtin_memcpy(&f, &x, 4); return f;
}
__device__ __forceinline__ unsigned int f2bfbits(float f) {
    __bf16 h = (__bf16)f;
    unsigned short u; __builtin_memcpy(&u, &h, 2); return (unsigned int)u;
}

// ---------------------------------------------------------------------------
// Plain WMMA GEMM:  C[M,Nout](bf16) = act(A)[M,KP](bf16, lda) @ W[K,Nout](f32)
// If PRE: act(a)_k = relu(preScale[k]*a + preShift[k])  (BN+ReLU fused on A).
// W column-slab [col0,col0+NF*16) staged transposed into LDS as bf16;
// preScale/preShift staged as f32 after the weights.
// 256 threads = 8 waves; wave w handles row tile (blockIdx.x*8+w)*16.
// ---------------------------------------------------------------------------
template <int NF, bool PRE>
__global__ void wmma_gemm_plain(const __bf16* __restrict__ A, int lda,
                                const float* __restrict__ W, int K, int KP,
                                int Nout, __bf16* __restrict__ C, int ldc,
                                int Mtiles,
                                const float* __restrict__ preScale,
                                const float* __restrict__ preShift)
{
    constexpr int NCOL = NF * 16;
    extern __shared__ char smem[];
    __bf16* wlds = (__bf16*)smem;                       // NCOL*KP bf16
    float* sl = (float*)(smem + (size_t)NCOL * KP * 2); // KP f32 (PRE)
    float* sh = sl + KP;                                // KP f32 (PRE)

    const int col0 = blockIdx.y * NCOL;
    for (int i = threadIdx.x; i < NCOL * KP; i += blockDim.x) {
        int nl = i / KP, k = i - nl * KP;
        float v = 0.f;
        if (k < K) v = W[(size_t)k * Nout + col0 + nl];
        wlds[i] = (__bf16)v;
    }
    if (PRE) {
        for (int i = threadIdx.x; i < KP; i += blockDim.x) {
            sl[i] = (i < K) ? preScale[i] : 0.f;
            sh[i] = (i < K) ? preShift[i] : 0.f;
        }
    }
    __syncthreads();

    const int wave = threadIdx.x >> 5, lane = threadIdx.x & 31;
    const int tile = blockIdx.x * 8 + wave;
    if (tile >= Mtiles) return;
    const int mBase = tile * 16;
    const int lo16 = lane & 15, hi = lane >> 4;
    const __bf16* arow = A + (size_t)(mBase + lo16) * lda;
    const int kA = hi * 8;    // A-fragment K offset (ISA 7.12.2, 16-bit A 16x32)
    const int kB = hi * 8;    // B-fragment K offset (same table order as A)

    v8f acc[NF] = {};
    Frag16 a, b;
    for (int kk = 0; kk < KP; kk += 32) {
        a.q[0] = *(const uint4*)(arow + kk + kA);
        a.q[1] = *(const uint4*)(arow + kk + kA + 16);
        if (PRE) {
            #pragma unroll
            for (int p = 0; p < 8; ++p) {
                int k = kk + kA + 2 * p + ((p >= 4) ? 8 : 0);
                float2 s2 = *(const float2*)(sl + k);
                float2 t2 = *(const float2*)(sh + k);
                float lo = fmaxf((float)a.h[2 * p]     * s2.x + t2.x, 0.f);
                float hh = fmaxf((float)a.h[2 * p + 1] * s2.y + t2.y, 0.f);
                a.h[2 * p]     = (__bf16)lo;
                a.h[2 * p + 1] = (__bf16)hh;
            }
        }
        #pragma unroll
        for (int j = 0; j < NF; ++j) {
            const uint4* bp = (const uint4*)(&wlds[(size_t)(j * 16 + lo16) * KP + kk + kB]);
            b.q[0] = bp[0]; b.q[1] = bp[2];
            acc[j] = __builtin_amdgcn_wmma_f32_16x16x32_bf16(
                false, a.v, false, b.v, (short)0, acc[j], false, false);
        }
    }
    #pragma unroll
    for (int j = 0; j < NF; ++j) {
        const int n = col0 + j * 16 + lo16;
        __bf16* cp = C + (size_t)(mBase + hi * 8) * ldc + n;
        #pragma unroll
        for (int r = 0; r < 8; ++r) cp[(size_t)r * ldc] = (__bf16)acc[j][r];
    }
}

// ---------------------------------------------------------------------------
// Edge-gather WMMA GEMM:  per edge e, feature row f = [x[dst], x[src]-x[dst]]
// (length 2*d, zero-padded to KP), C = F @ W. A-fragment built in-register
// from the bf16 node table X (L2-resident). EVEND: d even -> packed u32
// pair loads (pairs never straddle the concat boundary).
// ---------------------------------------------------------------------------
template <int NF, bool EVEND>
__global__ void wmma_gemm_edge(const __bf16* __restrict__ X, int ldx, int d,
                               const int* __restrict__ src, const int* __restrict__ dst,
                               const float* __restrict__ W, int KP, int Nout,
                               __bf16* __restrict__ C, int ldc, int Mtiles)
{
    constexpr int NCOL = NF * 16;
    extern __shared__ char smem[];
    __bf16* wlds = (__bf16*)smem;
    const int col0 = blockIdx.y * NCOL;
    const int K = 2 * d;

    for (int i = threadIdx.x; i < NCOL * KP; i += blockDim.x) {
        int nl = i / KP, k = i - nl * KP;
        float v = 0.f;
        if (k < K) v = W[(size_t)k * Nout + col0 + nl];
        wlds[i] = (__bf16)v;
    }
    __syncthreads();

    const int wave = threadIdx.x >> 5, lane = threadIdx.x & 31;
    const int tile = blockIdx.x * 8 + wave;
    if (tile >= Mtiles) return;
    const int mBase = tile * 16;
    const int lo16 = lane & 15, hi = lane >> 4;
    const int e = mBase + lo16;
    const size_t di = (size_t)dst[e] * ldx;
    const size_t si = (size_t)src[e] * ldx;
    const int kA = hi * 8, kB = hi * 8;

    v8f acc[NF] = {};
    Frag16 a, b;
    for (int kk = 0; kk < KP; kk += 32) {
        if (EVEND) {
            #pragma unroll
            for (int p = 0; p < 8; ++p) {
                int k = kk + kA + 2 * p + ((p >= 4) ? 8 : 0);
                unsigned int v = 0u;
                if (k < d) {
                    v = *(const unsigned int*)(X + di + k);          // xi pair
                } else if (k < K) {
                    int k2 = k - d;
                    unsigned int aj = *(const unsigned int*)(X + si + k2);
                    unsigned int ai = *(const unsigned int*)(X + di + k2);
                    float lo = bfbits2f(aj & 0xffffu) - bfbits2f(ai & 0xffffu);
                    float hh = bfbits2f(aj >> 16)     - bfbits2f(ai >> 16);
                    v = f2bfbits(lo) | (f2bfbits(hh) << 16);
                }
                a.u[p] = v;
            }
        } else {
            #pragma unroll
            for (int i = 0; i < 16; ++i) {
                int k = kk + kA + i + ((i >= 8) ? 8 : 0);
                float f = 0.f;
                if (k < d) {
                    f = (float)X[di + k];
                } else if (k < K) {
                    int k2 = k - d;
                    f = (float)X[si + k2] - (float)X[di + k2];
                }
                a.h[i] = (__bf16)f;
            }
        }
        #pragma unroll
        for (int j = 0; j < NF; ++j) {
            const uint4* bp = (const uint4*)(&wlds[(size_t)(j * 16 + lo16) * KP + kk + kB]);
            b.q[0] = bp[0]; b.q[1] = bp[2];
            acc[j] = __builtin_amdgcn_wmma_f32_16x16x32_bf16(
                false, a.v, false, b.v, (short)0, acc[j], false, false);
        }
    }
    #pragma unroll
    for (int j = 0; j < NF; ++j) {
        const int n = col0 + j * 16 + lo16;
        __bf16* cp = C + (size_t)(mBase + hi * 8) * ldc + n;
        #pragma unroll
        for (int r = 0; r < 8; ++r) cp[(size_t)r * ldc] = (__bf16)acc[j][r];
    }
}

// ---------------------------------------------------------------------------
// Column stats: blockDim.x == C (32/64/128/384). Thread t owns column t.
// ---------------------------------------------------------------------------
__global__ void col_stats(const __bf16* __restrict__ Y, int C, int M,
                          float* __restrict__ sum, float* __restrict__ ssq)
{
    const int c = threadIdx.x;
    float s = 0.f, q = 0.f;
    for (int r = blockIdx.x; r < M; r += gridDim.x) {
        float v = (float)Y[(size_t)r * C + c];
        s += v; q += v * v;
    }
    atomAddF(&sum[c], s);
    atomAddF(&ssq[c], q);
}

__global__ void bn_finalize(const float* __restrict__ sum, const float* __restrict__ ssq,
                            const float* __restrict__ g, const float* __restrict__ be,
                            float invM, int C,
                            float* __restrict__ scale, float* __restrict__ shift)
{
    int c = blockIdx.x * blockDim.x + threadIdx.x;
    if (c >= C) return;
    float m = sum[c] * invM;
    float var = ssq[c] * invM - m * m;
    float inv = rsqrtf(var + BNEPS);
    float sc = g[c] * inv;
    scale[c] = sc;
    shift[c] = be[c] - m * sc;
}

__global__ void edge_scatter(const __bf16* __restrict__ Y2, const int* __restrict__ dst,
                             const float* __restrict__ scale, const float* __restrict__ shift,
                             int C, long long n, float* __restrict__ accum)
{
    long long i = (long long)blockIdx.x * blockDim.x + threadIdx.x;
    if (i >= n) return;
    int c = (int)(i % C);
    int e = (int)(i / C);
    float v = fmaxf(scale[c] * (float)Y2[i] + shift[c], 0.f);
    atomAddF(&accum[(size_t)dst[e] * C + c], v);
}

__global__ void count_idx(const int* __restrict__ idx, float* __restrict__ cnt, int n)
{
    int i = blockIdx.x * blockDim.x + threadIdx.x;
    if (i < n) atomAddF(&cnt[idx[i]], 1.f);
}

__global__ void cast_x3(const float* __restrict__ x, __bf16* __restrict__ xb, int n)
{
    int i = blockIdx.x * blockDim.x + threadIdx.x;
    if (i >= n) return;
    int nn = i / 3, c = i - nn * 3;
    xb[(size_t)nn * 128 + c] = (__bf16)x[i];
}

__global__ void cast_f32_bf16(const float* __restrict__ a, __bf16* __restrict__ b, long long n)
{
    long long i = (long long)blockIdx.x * blockDim.x + threadIdx.x;
    if (i < n) b[i] = (__bf16)a[i];
}

// combine with shortcut linear: x = relu(BN(S) + accum/cnt)
__global__ void combine_sc(float* __restrict__ hc, const __bf16* __restrict__ S,
                           const float* __restrict__ scS, const float* __restrict__ shS,
                           const float* __restrict__ accum, const float* __restrict__ cnt,
                           __bf16* __restrict__ xb, int C, long long n)
{
    long long i = (long long)blockIdx.x * blockDim.x + threadIdx.x;
    if (i >= n) return;
    int c = (int)(i % C);
    int nn = (int)(i / C);
    float base = scS[c] * (float)S[i] + shS[c];
    float v = fmaxf(base + accum[i] / fmaxf(cnt[nn], 1.f), 0.f);
    hc[(size_t)nn * 448 + c] = v;
    xb[(size_t)nn * 128 + c] = (__bf16)v;
}

// combine with identity shortcut: x = relu(prev + accum/cnt)
__global__ void combine_prev(float* __restrict__ hc, const float* __restrict__ hp,
                             const float* __restrict__ accum, const float* __restrict__ cnt,
                             __bf16* __restrict__ xb, int C, long long n)
{
    long long i = (long long)blockIdx.x * blockDim.x + threadIdx.x;
    if (i >= n) return;
    int c = (int)(i % C);
    int nn = (int)(i / C);
    float v = fmaxf(hp[(size_t)nn * 448 + c] + accum[i] / fmaxf(cnt[nn], 1.f), 0.f);
    hc[(size_t)nn * 448 + c] = v;
    xb[(size_t)nn * 128 + c] = (__bf16)v;
}

__global__ void pool_kernel(const __bf16* __restrict__ H1, const int* __restrict__ batch,
                            const float* __restrict__ scale, const float* __restrict__ shift,
                            float* __restrict__ gp, long long n)
{
    long long i = (long long)blockIdx.x * blockDim.x + threadIdx.x;
    if (i >= n) return;
    int c = (int)(i % 384);
    int nn = (int)(i / 384);
    float v = fmaxf(scale[c] * (float)H1[i] + shift[c], 0.f);
    atomAddF(&gp[(size_t)batch[nn] * 384 + c], v);
}

// Head: seq2 (384->256) + relu + lin (256->1) + softmax over axis of size 1.
__global__ void head_kernel(const float* __restrict__ gp, const float* __restrict__ gcnt,
                            const float* __restrict__ w2, const float* __restrict__ b2,
                            const float* __restrict__ lw, const float* __restrict__ lb,
                            float* __restrict__ out)
{
    __shared__ float gb[384];
    __shared__ float red[256];
    const int b = blockIdx.x, t = threadIdx.x;
    float inv = 1.f / fmaxf(gcnt[b], 1.f);
    for (int i = t; i < 384; i += 256) gb[i] = gp[(size_t)b * 384 + i] * inv;
    __syncthreads();
    float s = b2[t];
    for (int k = 0; k < 384; ++k) s += gb[k] * w2[(size_t)k * 256 + t];
    red[t] = fmaxf(s, 0.f) * lw[t];
    __syncthreads();
    for (int off = 128; off > 0; off >>= 1) {
        if (t < off) red[t] += red[t + off];
        __syncthreads();
    }
    if (t == 0) {
        float o = red[0] + lb[0];
        float m = o;                       // max over softmax axis (length 1)
        float e = __expf(o - m);
        out[b] = e / e;                    // softmax over axis of size 1
    }
}

// ---------------------------------------------------------------------------
// Host orchestration
// ---------------------------------------------------------------------------
static inline int r32(int k) { return (k + 31) & ~31; }

extern "C" void kernel_launch(void* const* d_in, const int* in_sizes, int n_in,
                              void* d_out, int out_size, void* d_ws, size_t ws_size,
                              hipStream_t stream)
{
    (void)in_sizes; (void)n_in; (void)out_size; (void)ws_size;
    const float* x    = (const float*)d_in[0];
    const int* ei     = (const int*)d_in[1];
    const int* batch  = (const int*)d_in[2];
    const int* src = ei;
    const int* dst = ei + GE;

    // conv params: base index per layer; +0 w1, +2 g1, +3 be1, +4 w2, +6 g2, +7 be2
    const int convBase[6] = {3, 11, 19, 27, 35, 43};
    const int scBase[3]   = {51, 54, 57};          // sc1, sc3, sc5 (w,g,be)
    const float* seq1_w  = (const float*)d_in[60];
    const float* seq1_g  = (const float*)d_in[62];
    const float* seq1_be = (const float*)d_in[63];
    const float* seq2_w  = (const float*)d_in[64];
    const float* seq2_b  = (const float*)d_in[65];
    const float* lin_w   = (const float*)d_in[66];
    const float* lin_b   = (const float*)d_in[67];

    // workspace carve-up
    char* w = (char*)d_ws;
    auto carve = [&](size_t bytes) { char* p = w; w += (bytes + 255) & ~(size_t)255; return p; };
    __bf16* Y1   = (__bf16*)carve((size_t)GE * 128 * 2);
    __bf16* Y2   = (__bf16*)carve((size_t)GE * 128 * 2);
    float*  Hc   = (float*) carve((size_t)GN * 448 * 4);
    __bf16* Hb   = (__bf16*)carve((size_t)GN * 448 * 2);
    __bf16* Xb   = (__bf16*)carve((size_t)GN * 128 * 2);
    __bf16* S    = (__bf16*)carve((size_t)GN * 128 * 2);
    __bf16* H1   = (__bf16*)carve((size_t)GN * 384 * 2);
    float*  accum= (float*) carve((size_t)GN * 128 * 4);
    float*  cnt  = (float*) carve((size_t)GN * 4);
    float*  gp   = (float*) carve((size_t)GB * 384 * 4);
    float*  gcnt = (float*) carve((size_t)GB * 4);
    float*  st   = (float*) carve(3 * 4 * 512 * 4);
    auto sSum   = [&](int s_) { return st + s_ * 2048; };
    auto sSsq   = [&](int s_) { return st + s_ * 2048 + 512; };
    auto sScale = [&](int s_) { return st + s_ * 2048 + 1024; };
    auto sShift = [&](int s_) { return st + s_ * 2048 + 1536; };

    auto ew = [](long long n) { return dim3((unsigned)((n + 255) / 256)); };

    // GEMM dispatchers: NF chosen at compile time (2 for Nout=32, else 4).
    auto launch_plain = [&](const __bf16* A, int lda, const float* Wp, int K, int KP,
                            int Nout, __bf16* C, int ldc, int Mtiles,
                            const float* preSc, const float* preSh) {
        dim3 grid((Mtiles + 7) / 8, (Nout >= 64) ? Nout / 64 : 1);
        size_t sw = (size_t)((Nout >= 64) ? 64 : 32) * KP * 2;
        size_t sp = preSc ? (size_t)KP * 8 : 0;
        if (Nout >= 64) {
            if (preSc) wmma_gemm_plain<4, true ><<<grid, 256, sw + sp, stream>>>(A, lda, Wp, K, KP, Nout, C, ldc, Mtiles, preSc, preSh);
            else       wmma_gemm_plain<4, false><<<grid, 256, sw,      stream>>>(A, lda, Wp, K, KP, Nout, C, ldc, Mtiles, nullptr, nullptr);
        } else {
            if (preSc) wmma_gemm_plain<2, true ><<<grid, 256, sw + sp, stream>>>(A, lda, Wp, K, KP, Nout, C, ldc, Mtiles, preSc, preSh);
            else       wmma_gemm_plain<2, false><<<grid, 256, sw,      stream>>>(A, lda, Wp, K, KP, Nout, C, ldc, Mtiles, nullptr, nullptr);
        }
    };
    auto launch_edge = [&](const __bf16* X, int ldx, int d, const float* Wp, int KP,
                           int Nout, __bf16* C, int ldc, int Mtiles) {
        dim3 grid((Mtiles + 7) / 8, (Nout >= 64) ? Nout / 64 : 1);
        size_t sw = (size_t)((Nout >= 64) ? 64 : 32) * KP * 2;
        bool evend = (d % 2) == 0;
        if (Nout >= 64) {
            if (evend) wmma_gemm_edge<4, true ><<<grid, 256, sw, stream>>>(X, ldx, d, src, dst, Wp, KP, Nout, C, ldc, Mtiles);
            else       wmma_gemm_edge<4, false><<<grid, 256, sw, stream>>>(X, ldx, d, src, dst, Wp, KP, Nout, C, ldc, Mtiles);
        } else {
            if (evend) wmma_gemm_edge<2, true ><<<grid, 256, sw, stream>>>(X, ldx, d, src, dst, Wp, KP, Nout, C, ldc, Mtiles);
            else       wmma_gemm_edge<2, false><<<grid, 256, sw, stream>>>(X, ldx, d, src, dst, Wp, KP, Nout, C, ldc, Mtiles);
        }
    };

    // init
    hipMemsetAsync(Xb, 0, (size_t)GN * 128 * 2, stream);
    hipMemsetAsync(cnt, 0, (size_t)GN * 4, stream);
    cast_x3<<<ew(GN * 3), 256, 0, stream>>>(x, Xb, GN * 3);
    count_idx<<<ew(GE), 256, 0, stream>>>(dst, cnt, GE);

    const int dims[6][3] = {{3,32,32},{32,32,32},{32,64,64},{64,64,64},{64,128,128},{128,128,128}};
    const int colOff[6]  = {0, 32, 64, 128, 192, 320};
    const int prevOff[6] = {0, 0, 32, 64, 128, 192};
    const int Etiles = GE / 16;                  // 100000
    const int Ntiles = GN / 16;                  // 6250

    int scIdx = 0;
    for (int l = 0; l < 6; ++l) {
        const int d = dims[l][0], h = dims[l][1], o = dims[l][2];
        const int base = convBase[l];
        const float* w1  = (const float*)d_in[base + 0];
        const float* g1  = (const float*)d_in[base + 2];
        const float* be1 = (const float*)d_in[base + 3];
        const float* w2  = (const float*)d_in[base + 4];
        const float* g2  = (const float*)d_in[base + 6];
        const float* be2 = (const float*)d_in[base + 7];

        // --- GEMM1: edge-gathered F @ w1 -> Y1 [E,h] (raw, pre-BN) ---
        hipMemsetAsync(sSum(1), 0, 1024 * 4, stream);
        launch_edge(Xb, 128, d, w1, r32(2 * d), h, Y1, h, Etiles);
        col_stats<<<2048, h, 0, stream>>>(Y1, h, GE, sSum(1), sSsq(1));
        bn_finalize<<<(h + 63) / 64, 64, 0, stream>>>(sSum(1), sSsq(1), g1, be1,
                                                      1.f / GE, h, sScale(1), sShift(1));
        // --- GEMM2: relu(BN(Y1)) @ w2 -> Y2 [E,o]  (BN+ReLU fused on A) ---
        hipMemsetAsync(sSum(2), 0, 1024 * 4, stream);
        launch_plain(Y1, h, w2, h, h, o, Y2, o, Etiles, sScale(1), sShift(1));
        col_stats<<<2048, o, 0, stream>>>(Y2, o, GE, sSum(2), sSsq(2));
        bn_finalize<<<(o + 63) / 64, 64, 0, stream>>>(sSum(2), sSsq(2), g2, be2,
                                                      1.f / GE, o, sScale(2), sShift(2));
        // --- scatter-mean numerator ---
        hipMemsetAsync(accum, 0, (size_t)GN * o * 4, stream);
        edge_scatter<<<ew((long long)GE * o), 256, 0, stream>>>(
            Y2, dst, sScale(2), sShift(2), o, (long long)GE * o, accum);

        float* hcol = Hc + colOff[l];
        if ((l & 1) == 0) {
            // shortcut linear + BN
            const float* scw  = (const float*)d_in[scBase[scIdx] + 0];
            const float* scg  = (const float*)d_in[scBase[scIdx] + 1];
            const float* scbe = (const float*)d_in[scBase[scIdx] + 2];
            ++scIdx;
            hipMemsetAsync(sSum(0), 0, 1024 * 4, stream);
            launch_plain(Xb, 128, scw, d, r32(d), o, S, o, Ntiles, nullptr, nullptr);
            col_stats<<<2048, o, 0, stream>>>(S, o, GN, sSum(0), sSsq(0));
            bn_finalize<<<(o + 63) / 64, 64, 0, stream>>>(sSum(0), sSsq(0), scg, scbe,
                                                          1.f / GN, o, sScale(0), sShift(0));
            combine_sc<<<ew((long long)GN * o), 256, 0, stream>>>(
                hcol, S, sScale(0), sShift(0), accum, cnt, Xb, o, (long long)GN * o);
        } else {
            combine_prev<<<ew((long long)GN * o), 256, 0, stream>>>(
                hcol, Hc + prevOff[l], accum, cnt, Xb, o, (long long)GN * o);
        }
    }

    // --- seq1: Hc[N,448] @ seq1_w -> H1[N,384], BN+relu fused into pooling ---
    cast_f32_bf16<<<ew((long long)GN * 448), 256, 0, stream>>>(Hc, Hb, (long long)GN * 448);
    hipMemsetAsync(sSum(1), 0, 1024 * 4, stream);
    launch_plain(Hb, 448, seq1_w, 448, 448, 384, H1, 384, Ntiles, nullptr, nullptr);
    col_stats<<<2048, 384, 0, stream>>>(H1, 384, GN, sSum(1), sSsq(1));
    bn_finalize<<<(384 + 63) / 64, 64, 0, stream>>>(sSum(1), sSsq(1), seq1_g, seq1_be,
                                                    1.f / GN, 384, sScale(1), sShift(1));
    // --- global mean pool + head ---
    hipMemsetAsync(gp, 0, (size_t)GB * 384 * 4, stream);
    hipMemsetAsync(gcnt, 0, (size_t)GB * 4, stream);
    count_idx<<<ew(GN), 256, 0, stream>>>(batch, gcnt, GN);
    pool_kernel<<<ew((long long)GN * 384), 256, 0, stream>>>(
        H1, batch, sScale(1), sShift(1), gp, (long long)GN * 384);
    head_kernel<<<GB, 256, 0, stream>>>(gp, gcnt, seq2_w, seq2_b, lin_w, lin_b, (float*)d_out);
}
